// GrapherDGL_83777632076275
// MI455X (gfx1250) — matrix operand order repaired and checked by reading, not compiled, not run on hardware
//
#include <hip/hip_runtime.h>

typedef __attribute__((ext_vector_type(16))) _Float16 v16h;
typedef __attribute__((ext_vector_type(8)))  float    v8f;

#define BN_B 8
#define CH   128
#define NPTS 4096
#define KNB  16
#define NODES (BN_B * NPTS)   // 32768

union AF { v16h v; unsigned u[8]; };

// --- gfx1250 async global->LDS copy (ASYNCcnt-tracked), per CDNA5 ISA ---
__device__ __forceinline__ void async_to_lds_b128(unsigned lds_off, const void* g) {
  unsigned long long ga = (unsigned long long)g;
  asm volatile("global_load_async_to_lds_b128 %0, %1, off" :: "v"(lds_off), "v"(ga) : "memory");
}
__device__ __forceinline__ void wait_async_lds() {
  asm volatile("s_wait_asynccnt 0" ::: "memory");
}

__device__ __forceinline__ v8f wmma_f16(const AF& A, const AF& B, v8f C) {
  return __builtin_amdgcn_wmma_f32_16x16x32_f16(false, A.v, false, B.v, (short)0, C,
                                                false, false);
}

__device__ __forceinline__ v8f zero8() {
  v8f z;
  #pragma unroll
  for (int e = 0; e < 8; ++e) z[e] = 0.0f;
  return z;
}

// ---------------------------------------------------------------------------
// Kernel 0: fold BN constants, convert weights fp32 -> fp16
// ---------------------------------------------------------------------------
__global__ void __launch_bounds__(256) prep_kernel(
    const float* __restrict__ W1,  const float* __restrict__ b1,
    const float* __restrict__ g1,  const float* __restrict__ be1,
    const float* __restrict__ m1,  const float* __restrict__ v1,
    const float* __restrict__ Wmr, const float* __restrict__ W2,
    const float* __restrict__ b2,  const float* __restrict__ g2,
    const float* __restrict__ be2, const float* __restrict__ m2,
    const float* __restrict__ v2,
    _Float16* __restrict__ W1h, _Float16* __restrict__ Wmrh, _Float16* __restrict__ W2h,
    float* __restrict__ a1, float* __restrict__ c1,
    float* __restrict__ a2, float* __restrict__ c2)
{
  int tid = blockIdx.x * blockDim.x + threadIdx.x;
  int stride = gridDim.x * blockDim.x;
  for (int i = tid; i < CH * CH; i += stride)           W1h[i]  = (_Float16)W1[i];
  for (int i = tid; i < 2 * CH * 2 * CH; i += stride)   Wmrh[i] = (_Float16)Wmr[i];
  for (int i = tid; i < CH * 2 * CH; i += stride)       W2h[i]  = (_Float16)W2[i];
  for (int o = tid; o < CH; o += stride) {
    float s  = g1[o] * rsqrtf(v1[o] + 1e-5f);
    a1[o] = s;  c1[o] = b1[o] * s + be1[o] - m1[o] * s;
    float s2 = g2[o] * rsqrtf(v2[o] + 1e-5f);
    a2[o] = s2; c2[o] = b2[o] * s2 + be2[o] - m2[o] * s2;
  }
}

// ---------------------------------------------------------------------------
// Kernel 1: fc1  h = ReLU(BN(W1 @ x + b1)), written node-major f16 into cat[:, 0:128]
// grid: NODES/128 blocks, 256 thr (8 waves); wave -> 16-node M tile, full 128 out-ch
// ---------------------------------------------------------------------------
__global__ void __launch_bounds__(256) fc1_kernel(
    const float* __restrict__ x, const _Float16* __restrict__ W1h,
    const float* __restrict__ a1, const float* __restrict__ c1,
    _Float16* __restrict__ cat)
{
  extern __shared__ char smem[];               // 32 KB: W1 in f16
  const int tid = threadIdx.x;
  unsigned lds_base = (unsigned)(uintptr_t)smem;
  #pragma unroll
  for (int r = 0; r < 8; ++r) {                // 2048 x 16B chunks / 256 threads
    int chunk = r * 256 + tid;
    async_to_lds_b128(lds_base + chunk * 16, (const char*)W1h + chunk * 16);
  }
  wait_async_lds();
  __syncthreads();

  const unsigned* wl = (const unsigned*)smem;  // row o: 64 dwords (128 halfs)
  const int wave = tid >> 5, lane = tid & 31;
  const int col = lane & 15, hs = lane >> 4;
  const int i0 = blockIdx.x * 128 + wave * 16;

  v8f acc[8];
  #pragma unroll
  for (int t = 0; t < 8; ++t) acc[t] = zero8();

  const int iA = i0 + col;                     // A row this lane supplies
  const size_t xbase = (size_t)(iA >> 12) * (CH * NPTS) + (size_t)(iA & (NPTS - 1));

  #pragma unroll
  for (int k0 = 0; k0 < CH; k0 += 32) {
    AF A;
    #pragma unroll
    for (int j = 0; j < 8; ++j) {              // A: fp32 gather (stride N) -> f16 pack
      int kk = k0 + (hs << 4) + (j << 1);
      A.v[2 * j]     = (_Float16)x[xbase + (size_t)kk * NPTS];
      A.v[2 * j + 1] = (_Float16)x[xbase + (size_t)(kk + 1) * NPTS];
    }
    #pragma unroll
    for (int nt = 0; nt < 8; ++nt) {
      AF Bm;
      int o  = nt * 16 + col;
      int dw = o * 64 + ((k0 + (hs << 4)) >> 1);
      #pragma unroll
      for (int j = 0; j < 8; ++j) Bm.u[j] = wl[dw + j];
      acc[nt] = wmma_f16(A, Bm, acc[nt]);
    }
  }

  #pragma unroll
  for (int nt = 0; nt < 8; ++nt) {
    int o = nt * 16 + col;
    float s = a1[o], t = c1[o];
    #pragma unroll
    for (int j = 0; j < 8; ++j) {
      int i = i0 + hs * 8 + j;
      float v = acc[nt][j] * s + t;
      v = v > 0.0f ? v : 0.0f;
      cat[(size_t)i * 256 + o] = (_Float16)v;
    }
  }
}

// ---------------------------------------------------------------------------
// Kernel 2: MRConv gather-max: cat[:,128:256] = max_k (xn[i] - xn[src[i,k]])
// 32 threads/node, 4 channels/thread; 8 nodes per 256-thr block
// ---------------------------------------------------------------------------
__global__ void __launch_bounds__(256) mrconv_kernel(
    const int* __restrict__ src, _Float16* __restrict__ cat)
{
  const int t = threadIdx.x;
  const int node = blockIdx.x * 8 + (t >> 5);
  const int cg = (t & 31) * 4;
  union Q { uint2 q; _Float16 h[4]; };

  Q d; d.q = *(const uint2*)(cat + (size_t)node * 256 + cg);
  float d0 = (float)d.h[0], d1 = (float)d.h[1], d2 = (float)d.h[2], d3 = (float)d.h[3];
  float m0 = -1e30f, m1 = -1e30f, m2 = -1e30f, m3 = -1e30f;

  const int* sp = src + (size_t)node * KNB;
  #pragma unroll 4
  for (int k = 0; k < KNB; ++k) {
    int s = sp[k];
    Q sv; sv.q = *(const uint2*)(cat + (size_t)s * 256 + cg);
    m0 = fmaxf(m0, d0 - (float)sv.h[0]);
    m1 = fmaxf(m1, d1 - (float)sv.h[1]);
    m2 = fmaxf(m2, d2 - (float)sv.h[2]);
    m3 = fmaxf(m3, d3 - (float)sv.h[3]);
  }
  Q o;
  o.h[0] = (_Float16)m0; o.h[1] = (_Float16)m1;
  o.h[2] = (_Float16)m2; o.h[3] = (_Float16)m3;
  *(uint2*)(cat + (size_t)node * 256 + 128 + cg) = o.q;
}

// ---------------------------------------------------------------------------
// Kernel 3: hn = ReLU(cat @ Wmr^T + bmr)   M=32768 K=256 N=256
// 8 waves: (wave&3) -> 32-row M block (2 WMMA tiles), (wave>>2) -> N half.
// Each LDS B tile feeds 2 WMMAs (halves LDS bytes per wmma).
// ---------------------------------------------------------------------------
__global__ void __launch_bounds__(256) fcmr_kernel(
    const _Float16* __restrict__ cat, const _Float16* __restrict__ Wmrh,
    const float* __restrict__ bmr, _Float16* __restrict__ hn)
{
  extern __shared__ char smem[];               // 128 KB
  const int tid = threadIdx.x;
  unsigned lds_base = (unsigned)(uintptr_t)smem;
  #pragma unroll
  for (int r = 0; r < 32; ++r) {               // 8192 x 16B chunks / 256 threads
    int chunk = r * 256 + tid;
    async_to_lds_b128(lds_base + chunk * 16, (const char*)Wmrh + chunk * 16);
  }
  wait_async_lds();
  __syncthreads();

  const unsigned* wl = (const unsigned*)smem;  // row o: 128 dwords (256 halfs)
  const int wave = tid >> 5, lane = tid & 31;
  const int col = lane & 15, hs = lane >> 4;
  const int i0 = blockIdx.x * 128 + (wave & 3) * 32;
  const int ob = (wave >> 2) * 128;
  const _Float16* arow0 = cat + (size_t)(i0 + col) * 256;
  const _Float16* arow1 = cat + (size_t)(i0 + 16 + col) * 256;

  v8f acc0[8], acc1[8];
  #pragma unroll
  for (int t = 0; t < 8; ++t) { acc0[t] = zero8(); acc1[t] = zero8(); }

  #pragma unroll
  for (int k0 = 0; k0 < 256; k0 += 32) {
    AF A0, A1;
    #pragma unroll
    for (int j = 0; j < 8; ++j) {
      int kk = k0 + (hs << 4) + (j << 1);
      A0.u[j] = *(const unsigned*)(arow0 + kk);
      A1.u[j] = *(const unsigned*)(arow1 + kk);
    }
    #pragma unroll
    for (int nt = 0; nt < 8; ++nt) {
      AF Bm;
      int o  = ob + nt * 16 + col;
      int dw = o * 128 + ((k0 + (hs << 4)) >> 1);
      #pragma unroll
      for (int j = 0; j < 8; ++j) Bm.u[j] = wl[dw + j];
      acc0[nt] = wmma_f16(A0, Bm, acc0[nt]);   // B reused for both M tiles
      acc1[nt] = wmma_f16(A1, Bm, acc1[nt]);
    }
  }

  #pragma unroll
  for (int nt = 0; nt < 8; ++nt) {
    int o = ob + nt * 16 + col;
    float bias = bmr[o];
    #pragma unroll
    for (int j = 0; j < 8; ++j) {
      int ia = i0 + hs * 8 + j;
      float v0 = acc0[nt][j] + bias;
      v0 = v0 > 0.0f ? v0 : 0.0f;
      hn[(size_t)ia * 256 + o] = (_Float16)v0;
      int ib = i0 + 16 + hs * 8 + j;
      float v1 = acc1[nt][j] + bias;
      v1 = v1 > 0.0f ? v1 : 0.0f;
      hn[(size_t)ib * 256 + o] = (_Float16)v1;
    }
  }
}

// ---------------------------------------------------------------------------
// Kernel 4: out = ReLU(BN(W2 @ hn) + x)   M=32768 K=256 N=128, [B,C,N] output
// wave -> 32-row M block (2 WMMA tiles) x full 128 out-ch; B tile reused 2x.
// ---------------------------------------------------------------------------
__global__ void __launch_bounds__(256) fc2_kernel(
    const _Float16* __restrict__ hn, const _Float16* __restrict__ W2h,
    const float* __restrict__ a2, const float* __restrict__ c2,
    const float* __restrict__ x, float* __restrict__ out)
{
  extern __shared__ char smem[];               // 64 KB
  const int tid = threadIdx.x;
  unsigned lds_base = (unsigned)(uintptr_t)smem;
  #pragma unroll
  for (int r = 0; r < 16; ++r) {               // 4096 x 16B chunks / 256 threads
    int chunk = r * 256 + tid;
    async_to_lds_b128(lds_base + chunk * 16, (const char*)W2h + chunk * 16);
  }
  wait_async_lds();
  __syncthreads();

  const unsigned* wl = (const unsigned*)smem;  // row o: 128 dwords (256 halfs)
  const int wave = tid >> 5, lane = tid & 31;
  const int col = lane & 15, hs = lane >> 4;
  const int i0 = blockIdx.x * 256 + wave * 32;
  const _Float16* arow0 = hn + (size_t)(i0 + col) * 256;
  const _Float16* arow1 = hn + (size_t)(i0 + 16 + col) * 256;

  v8f acc0[8], acc1[8];
  #pragma unroll
  for (int t = 0; t < 8; ++t) { acc0[t] = zero8(); acc1[t] = zero8(); }

  #pragma unroll
  for (int k0 = 0; k0 < 256; k0 += 32) {
    AF A0, A1;
    #pragma unroll
    for (int j = 0; j < 8; ++j) {
      int kk = k0 + (hs << 4) + (j << 1);
      A0.u[j] = *(const unsigned*)(arow0 + kk);
      A1.u[j] = *(const unsigned*)(arow1 + kk);
    }
    #pragma unroll
    for (int nt = 0; nt < 8; ++nt) {
      AF Bm;
      int o  = nt * 16 + col;
      int dw = o * 128 + ((k0 + (hs << 4)) >> 1);
      #pragma unroll
      for (int j = 0; j < 8; ++j) Bm.u[j] = wl[dw + j];
      acc0[nt] = wmma_f16(A0, Bm, acc0[nt]);
      acc1[nt] = wmma_f16(A1, Bm, acc1[nt]);
    }
  }

  #pragma unroll
  for (int nt = 0; nt < 8; ++nt) {
    int o = nt * 16 + col;
    float s = a2[o], t = c2[o];
    #pragma unroll
    for (int j = 0; j < 8; ++j) {
      {
        int i = i0 + hs * 8 + j;
        int b = i >> 12, n = i & (NPTS - 1);
        size_t off = (size_t)b * (CH * NPTS) + (size_t)o * NPTS + n;
        float v = acc0[nt][j] * s + t + x[off];
        out[off] = v > 0.0f ? v : 0.0f;
      }
      {
        int i = i0 + 16 + hs * 8 + j;
        int b = i >> 12, n = i & (NPTS - 1);
        size_t off = (size_t)b * (CH * NPTS) + (size_t)o * NPTS + n;
        float v = acc1[nt][j] * s + t + x[off];
        out[off] = v > 0.0f ? v : 0.0f;
      }
    }
  }
}

// ---------------------------------------------------------------------------
extern "C" void kernel_launch(void* const* d_in, const int* in_sizes, int n_in,
                              void* d_out, int out_size, void* d_ws, size_t ws_size,
                              hipStream_t stream) {
  const float* x   = (const float*)d_in[0];
  const int*   src = (const int*)  d_in[1];
  const float* W1  = (const float*)d_in[2];
  const float* b1  = (const float*)d_in[3];
  const float* g1  = (const float*)d_in[4];
  const float* be1 = (const float*)d_in[5];
  const float* m1  = (const float*)d_in[6];
  const float* v1  = (const float*)d_in[7];
  const float* Wmr = (const float*)d_in[8];
  const float* bmr = (const float*)d_in[9];
  const float* W2  = (const float*)d_in[10];
  const float* b2  = (const float*)d_in[11];
  const float* g2  = (const float*)d_in[12];
  const float* be2 = (const float*)d_in[13];
  const float* m2  = (const float*)d_in[14];
  const float* v2  = (const float*)d_in[15];

  char* ws = (char*)d_ws;
  _Float16* W1h  = (_Float16*)(ws);                 //  32 KB
  _Float16* W2h  = (_Float16*)(ws + 32768);         //  64 KB
  _Float16* Wmrh = (_Float16*)(ws + 98304);         // 128 KB
  float* a1 = (float*)(ws + 229376);
  float* c1 = a1 + CH;
  float* a2 = c1 + CH;
  float* c2 = a2 + CH;
  _Float16* cat = (_Float16*)(ws + 262144);               // 16 MB [NODES,256]
  _Float16* hn  = (_Float16*)(ws + 262144 + 16777216);    // 16 MB [NODES,256]

  prep_kernel<<<64, 256, 0, stream>>>(W1, b1, g1, be1, m1, v1, Wmr, W2, b2, g2, be2,
                                      m2, v2, W1h, Wmrh, W2h, a1, c1, a2, c2);
  fc1_kernel  <<<NODES / 128, 256, 32768,  stream>>>(x, W1h, a1, c1, cat);
  mrconv_kernel<<<NODES / 8,  256, 0,      stream>>>(src, cat);
  fcmr_kernel <<<NODES / 128, 256, 131072, stream>>>(cat, Wmrh, bmr, hn);
  fc2_kernel  <<<NODES / 256, 256, 65536,  stream>>>(hn, W2h, a2, c2, x, (float*)d_out);
}
